// S4Fourier_26328149525106
// MI455X (gfx1250) — compile-verified, matrix operand order checked by reading, not run on hardware
//
#include <hip/hip_runtime.h>

// S4 forward via chunked state-space scan (exactly equivalent to the FFT conv):
//   y[b,l] = sum_{t<=l} (C Ã^t B̃) x[b,l-t],  Ã,B̃ = bilinear discretization.
// Chunks of T=64: intra-chunk causal conv done with v_wmma_f32_16x16x32_bf16,
// inter-chunk state carried by a tiny 32-step scan + rank-64 correction.
//
// Workspace layout (floats): At 4096 | Bt 16384 | Apow 65*4096 | M 64*64*256 |
// CA 64*256*64 | hend 65536 | hin 65536 | Ktile (bf16) 64*256*256  => ~18.4 MB.

#define BSZ  32
#define LSEQ 2048
#define DIMC 256     // input channels D
#define NST  64      // state size N
#define ODIM 256     // output channels O
#define TCH  64      // chunk length
#define NCH  32      // number of chunks (LSEQ/TCH)
#define DTC  0.1f

typedef __attribute__((ext_vector_type(16))) __bf16        v16bf;
typedef __attribute__((ext_vector_type(8)))  float         v8f;
typedef __attribute__((ext_vector_type(4)))  unsigned int  v4u;

union Frag { v4u u[2]; v16bf v; };

__device__ __forceinline__ unsigned short f2bf(float f) {
  unsigned int u = __float_as_uint(f);
  u += 0x7FFFu + ((u >> 16) & 1u);          // round-to-nearest-even
  return (unsigned short)(u >> 16);
}

// ---------------------------------------------------------------------------
// Kernel 1: bilinear discretization. Gauss-Jordan inverse of (I + A*DT/2)
// (strongly diagonally dominant -> no pivoting), then
//   At = inv @ (I - A*DT/2)   [64x64],   Bt = DT * inv @ B   [64x256]
// ---------------------------------------------------------------------------
__global__ __launch_bounds__(256) void k_disc(const float* __restrict__ A,
                                              const float* __restrict__ B,
                                              float* __restrict__ At,
                                              float* __restrict__ Bt) {
  __shared__ float G[64][129];               // augmented [M | I]
  const int tid = threadIdx.x;
  const int r   = tid >> 2;                  // row 0..63
  const int c0  = (tid & 3) * 32;            // 4 thread-groups cover 128 cols

  for (int cc = 0; cc < 32; ++cc) {
    int c = c0 + cc;
    float v;
    if (c < 64) v = ((r == c) ? 1.0f : 0.0f) + 0.5f * DTC * A[r * 64 + c];
    else        v = ((c - 64) == r) ? 1.0f : 0.0f;
    G[r][c] = v;
  }
  __syncthreads();

  for (int k = 0; k < 64; ++k) {
    float pinv = 1.0f / G[k][k];
    __syncthreads();
    if (r == k)
      for (int cc = 0; cc < 32; ++cc) G[k][c0 + cc] *= pinv;
    __syncthreads();
    float f = G[r][k];
    __syncthreads();
    if (r != k)
      for (int cc = 0; cc < 32; ++cc) G[r][c0 + cc] -= f * G[k][c0 + cc];
    __syncthreads();
  }

  // A_tilde = Ainv @ (I - A*DT/2)   (Ainv = right half of G)
  for (int s = 0; s < 16; ++s) {
    int o = tid * 16 + s;
    int ro = o >> 6, co = o & 63;
    float acc = 0.f;
    for (int m = 0; m < 64; ++m) {
      float rhs = ((m == co) ? 1.0f : 0.0f) - 0.5f * DTC * A[m * 64 + co];
      acc += G[ro][64 + m] * rhs;
    }
    At[o] = acc;
  }
  // B_tilde = DT * Ainv @ B
  for (int s = 0; s < 64; ++s) {
    int o = tid + s * 256;
    int ro = o >> 8, io = o & 255;
    float acc = 0.f;
    for (int m = 0; m < 64; ++m) acc += G[ro][64 + m] * B[m * 256 + io];
    Bt[o] = DTC * acc;
  }
}

// ---------------------------------------------------------------------------
// Kernel 2: Apow[t] = A_tilde^t for t=0..64 (sequential, one block, LDS)
// ---------------------------------------------------------------------------
__global__ __launch_bounds__(256) void k_apow(const float* __restrict__ At,
                                              float* __restrict__ Apow) {
  __shared__ float Atl[64][65];
  __shared__ float cur[64][65];
  const int tid = threadIdx.x;
  for (int s = 0; s < 16; ++s) {
    int o = tid * 16 + s; int rr = o >> 6, cc = o & 63;
    Atl[rr][cc] = At[o];
    float id = (rr == cc) ? 1.0f : 0.0f;
    cur[rr][cc] = id;
    Apow[o] = id;
  }
  __syncthreads();
  for (int t = 1; t <= TCH; ++t) {
    float v[16];
    for (int s = 0; s < 16; ++s) {
      int o = tid * 16 + s; int rr = o >> 6, cc = o & 63;
      float acc = 0.f;
      for (int m = 0; m < 64; ++m) acc += Atl[rr][m] * cur[m][cc];
      v[s] = acc;
    }
    __syncthreads();
    for (int s = 0; s < 16; ++s) {
      int o = tid * 16 + s; int rr = o >> 6, cc = o & 63;
      cur[rr][cc] = v[s];
      Apow[(size_t)t * 4096 + o] = v[s];
    }
    __syncthreads();
  }
}

// ---------------------------------------------------------------------------
// Kernel 3 (one block per t): M[t] = Ã^t B̃ ; CA[t] = C Ã^{t+1} ;
// K[t] = C M[t] scattered into bf16 WMMA B-fragment tiles:
//   Ktile[((t*8+ic)*16+ot)*512 + lane*16 + e] = K[t][o][i]
//   with i = ic*32 + k(lane,e), o = ot*16 + (lane&15),
//   k = (e&7) + ((e>>3)<<4) + ((lane>>4)<<3)   (mirror of the 16-bit A layout)
// ---------------------------------------------------------------------------
__global__ __launch_bounds__(256) void k_mk(const float* __restrict__ Apow,
                                            const float* __restrict__ Bt,
                                            const float* __restrict__ C,
                                            float* __restrict__ M,
                                            float* __restrict__ CA,
                                            unsigned short* __restrict__ Ktile) {
  const int t   = blockIdx.x;
  const int tid = threadIdx.x;
  __shared__ float Ap [64][65];
  __shared__ float Ap1[64][65];
  __shared__ float Mt [64][257];
  for (int s = 0; s < 16; ++s) {
    int o = tid * 16 + s; int rr = o >> 6, cc = o & 63;
    Ap [rr][cc] = Apow[(size_t)t * 4096 + o];
    Ap1[rr][cc] = Apow[(size_t)(t + 1) * 4096 + o];
  }
  __syncthreads();
  // Mt = Ap @ Bt  (64 x 256)
  for (int s = 0; s < 64; ++s) {
    int o = tid + s * 256; int n = o >> 8, i = o & 255;
    float acc = 0.f;
    for (int m = 0; m < 64; ++m) acc += Ap[n][m] * Bt[m * 256 + i];
    Mt[n][i] = acc;
    M[(size_t)t * 16384 + o] = acc;
  }
  // CA[t] = C @ Ap1  (256 x 64)
  for (int s = 0; s < 64; ++s) {
    int o = tid + s * 256; int oo = o >> 6, nn = o & 63;
    float acc = 0.f;
    for (int m = 0; m < 64; ++m) acc += C[oo * 64 + m] * Ap1[m][nn];
    CA[(size_t)t * 16384 + o] = acc;
  }
  __syncthreads();
  // K[t][o][i] = C @ Mt -> bf16 fragment scatter
  for (int s = 0; s < 256; ++s) {
    int idx = tid + s * 256; int o = idx >> 8, i = idx & 255;
    float acc = 0.f;
    for (int n = 0; n < 64; ++n) acc += C[o * 64 + n] * Mt[n][i];
    int ic = i >> 5, kk = i & 31, ot = o >> 4;
    int L  = (o & 15) + (((kk >> 3) & 1) << 4);
    int e  = (kk & 7) + ((kk >> 4) << 3);
    Ktile[(((size_t)t * 8 + ic) * 16 + ot) * 512 + L * 16 + e] = f2bf(acc);
  }
}

// ---------------------------------------------------------------------------
// Kernel 4: per-chunk end-state from local inputs:
//   hend[b,c,n] = sum_t M[T-1-t][n,:] . x[b, c*T+t, :]
// ---------------------------------------------------------------------------
__global__ __launch_bounds__(64) void k_hend(const float* __restrict__ x,
                                             const float* __restrict__ M,
                                             float* __restrict__ hend) {
  const int blk = blockIdx.x;                // b*NCH + c
  const int b = blk >> 5, c = blk & 31;
  const int tid = threadIdx.x;               // state index n
  __shared__ float xs[TCH][DIMC];
  const float* xc = x + ((size_t)b * LSEQ + c * TCH) * DIMC;
  for (int e = tid; e < TCH * DIMC; e += 64) xs[e >> 8][e & 255] = xc[e];
  __syncthreads();
  float acc = 0.f;
  for (int t = 0; t < TCH; ++t) {
    const float* Mr = M + ((size_t)(TCH - 1 - t) * 64 + tid) * 256;
    for (int i = 0; i < DIMC; ++i) acc += Mr[i] * xs[t][i];
  }
  hend[(size_t)blk * 64 + tid] = acc;
}

// ---------------------------------------------------------------------------
// Kernel 5: 32-step inter-chunk scan (per batch):
//   hin[b,0]=0 ; hin[b,c] = Ã^T hin[b,c-1] + hend[b,c-1]
// ---------------------------------------------------------------------------
__global__ __launch_bounds__(64) void k_scan(const float* __restrict__ Apow,
                                             const float* __restrict__ hend,
                                             float* __restrict__ hin) {
  const int b = blockIdx.x;
  const int n = threadIdx.x;
  __shared__ float h[64];
  const float* AT = Apow + (size_t)TCH * 4096;   // Ã^T
  h[n] = 0.f;
  hin[((size_t)b * NCH) * 64 + n] = 0.f;
  __syncthreads();
  for (int c = 1; c < NCH; ++c) {
    float acc = hend[((size_t)b * NCH + c - 1) * 64 + n];
    for (int m = 0; m < 64; ++m) acc += AT[n * 64 + m] * h[m];
    __syncthreads();
    h[n] = acc;
    hin[((size_t)b * NCH + c) * 64 + n] = acc;
    __syncthreads();
  }
}

// ---------------------------------------------------------------------------
// Kernel 6 (hot loop, WMMA): intra-chunk causal conv, one block per (b,c).
// Y[j,o] = sum_{t<=j} x[j-t,:] . K[t][o,:]   via 16x16x32 bf16 WMMA, f32 acc.
// Loop nest: t outer, j-tiles inner -> each Ktile B-fragment is loaded from
// global/L2 exactly once per (t,ic) and reused across all valid j-tiles
// (jt >= t>>4), cutting VMEM traffic 2.5x vs jt-outer. 16 leading zero rows
// in LDS make the j<t causal boundary contribute zero for free. Every wave
// executes an identical, perfectly balanced 2560-WMMA schedule.
// ---------------------------------------------------------------------------
__global__ __launch_bounds__(256) void k_conv(const float* __restrict__ x,
                                              const unsigned short* __restrict__ Ktile,
                                              float* __restrict__ y) {
  __shared__ __align__(16) unsigned short Xpad[(16 + TCH) * DIMC]; // 40 KB
  const int blk = blockIdx.x;                // b*NCH + c
  const int b = blk >> 5, c = blk & 31;
  const int tid = threadIdx.x;

  for (int e = tid; e < 16 * DIMC; e += 256) Xpad[e] = 0;          // zero pad
  const float* xc = x + ((size_t)b * LSEQ + c * TCH) * DIMC;
  for (int e = tid; e < TCH * DIMC; e += 256) Xpad[16 * DIMC + e] = f2bf(xc[e]);
  __syncthreads();

  const int w    = tid >> 5;                 // wave 0..7 -> o-tiles {2w, 2w+1}
  const int lane = tid & 31;
  const int m    = lane & 15;
  const int kb   = (lane >> 4) << 3;         // A-fragment k-base: 0 or 8

  v8f acc[8];
  #pragma unroll
  for (int q = 0; q < 8; ++q)
    #pragma unroll
    for (int r = 0; r < 8; ++r) acc[q][r] = 0.f;

  #pragma unroll
  for (int phase = 0; phase < 4; ++phase) {  // phase = t>>4 = min valid jt
    for (int tt = 0; tt < 16; ++tt) {
      const int t = phase * 16 + tt;
      #pragma unroll
      for (int ic = 0; ic < 8; ++ic) {
        // B fragments for (t, ic): loaded once, reused across j-tiles
        const unsigned short* bp =
            Ktile + (((size_t)t * 8 + ic) * 16 + w * 2) * 512 + lane * 16;
        Frag b0, b1;
        b0.u[0] = *(const v4u*)(bp);
        b0.u[1] = *(const v4u*)(bp + 8);
        b1.u[0] = *(const v4u*)(bp + 512);
        b1.u[1] = *(const v4u*)(bp + 520);
        #pragma unroll
        for (int jt = phase; jt < 4; ++jt) {
          const unsigned short* arp =
              &Xpad[(16 + jt * 16 - t + m) * DIMC + ic * 32 + kb];
          Frag a;
          a.u[0] = *(const v4u*)(arp);        // k = kb..kb+7
          a.u[1] = *(const v4u*)(arp + 16);   // k = kb+16..kb+23
          acc[jt * 2 + 0] = __builtin_amdgcn_wmma_f32_16x16x32_bf16(
              false, a.v, false, b0.v, (short)0, acc[jt * 2 + 0], false, false);
          acc[jt * 2 + 1] = __builtin_amdgcn_wmma_f32_16x16x32_bf16(
              false, a.v, false, b1.v, (short)0, acc[jt * 2 + 1], false, false);
        }
      }
    }
  }

  // D layout: VGPR r -> row m = r (+8 for lanes 16..31), col n = lane&15
  const int n   = lane & 15;
  const int rhi = (lane >> 4) << 3;
  #pragma unroll
  for (int jt = 0; jt < 4; ++jt) {
    #pragma unroll
    for (int q = 0; q < 2; ++q) {
      float* yb = y + ((size_t)b * LSEQ + c * TCH + jt * 16 + rhi) * DIMC
                    + (w * 2 + q) * 16 + n;
      const v8f av = acc[jt * 2 + q];
      #pragma unroll
      for (int r = 0; r < 8; ++r) yb[(size_t)r * DIMC] = av[r];
    }
  }
}

// ---------------------------------------------------------------------------
// Kernel 7: carry-in correction: y[b,cT+j,o] += CA[j][o,:] . hin[b,c,:]
// One block per (c,j); each thread register-caches its CA row once and loops
// over all 32 batches -> CA is read 32x (once per chunk c) instead of ~1000x.
// ---------------------------------------------------------------------------
__global__ __launch_bounds__(256) void k_correct(const float* __restrict__ CA,
                                                 const float* __restrict__ hin,
                                                 float* __restrict__ y) {
  const int blk = blockIdx.x;                // c*TCH + j
  const int j = blk & 63;
  const int c = blk >> 6;
  const int o = threadIdx.x;
  __shared__ float hs[BSZ][NST];             // 8 KB: hin for all batches
  for (int e = o; e < BSZ * NST; e += 256) {
    int bb = e >> 6, nn = e & 63;
    hs[bb][nn] = hin[((size_t)bb * NCH + c) * 64 + nn];
  }
  __syncthreads();
  float car[64];
  const float* cap = CA + ((size_t)j * 256 + o) * 64;
  #pragma unroll
  for (int nn = 0; nn < 64; ++nn) car[nn] = cap[nn];
  for (int bb = 0; bb < BSZ; ++bb) {
    float acc = 0.f;
    #pragma unroll
    for (int nn = 0; nn < 64; ++nn) acc += car[nn] * hs[bb][nn];
    y[((size_t)bb * LSEQ + c * TCH + j) * DIMC + o] += acc;
  }
}

// ---------------------------------------------------------------------------
extern "C" void kernel_launch(void* const* d_in, const int* in_sizes, int n_in,
                              void* d_out, int out_size, void* d_ws, size_t ws_size,
                              hipStream_t stream) {
  (void)in_sizes; (void)n_in; (void)out_size; (void)ws_size;
  const float* x = (const float*)d_in[0];    // [32,2048,256]
  const float* A = (const float*)d_in[1];    // [64,64]
  const float* B = (const float*)d_in[2];    // [64,256]
  const float* C = (const float*)d_in[3];    // [256,64]
  float* y  = (float*)d_out;                 // [32,2048,256]
  float* ws = (float*)d_ws;

  float* At   = ws;                          // 4096
  float* Bt   = ws + 4096;                   // 16384
  float* Apow = ws + 20480;                  // 65*4096
  float* M    = ws + 286720;                 // 64*64*256
  float* CA   = ws + 1335296;                // 64*256*64
  float* hend = ws + 2383872;                // 32*32*64
  float* hin  = ws + 2449408;                // 32*32*64
  unsigned short* Ktile = (unsigned short*)(ws + 2514944); // 64*256*256 bf16
  // total workspace: 2514944*4 + 8388608 = ~18.4 MB

  k_disc   <<<1,          256, 0, stream>>>(A, B, At, Bt);
  k_apow   <<<1,          256, 0, stream>>>(At, Apow);
  k_mk     <<<TCH,        256, 0, stream>>>(Apow, Bt, C, M, CA, Ktile);
  k_hend   <<<BSZ * NCH,  64,  0, stream>>>(x, M, hend);
  k_scan   <<<BSZ,        64,  0, stream>>>(Apow, hend, hin);
  k_conv   <<<BSZ * NCH,  256, 0, stream>>>(x, Ktile, y);
  k_correct<<<NCH * TCH,  256, 0, stream>>>(CA, hin, y);
}
